// ODEfunc_26053271618194
// MI455X (gfx1250) — matrix operand order repaired and checked by reading, not compiled, not run on hardware
//
#include <hip/hip_runtime.h>

typedef __attribute__((ext_vector_type(16))) _Float16 v16h;
typedef __attribute__((ext_vector_type(2)))  __fp16   v2h;   // matches __builtin_amdgcn_cvt_pkrtz return type
typedef __attribute__((ext_vector_type(8)))  float    v8f;
typedef __attribute__((ext_vector_type(4)))  unsigned int v4u;
typedef __attribute__((ext_vector_type(8)))  unsigned int v8u;

#define HID 64
#define BATCH 1048576
#define ROWS_PER_TILE 32
#define NTILES (BATCH / ROWS_PER_TILE)          // 32768
#define BLOCK 128
#define WAVES_PER_BLOCK (BLOCK / 32)
#define GRID 1024
#define TOTAL_WAVES (GRID * WAVES_PER_BLOCK)    // 4096 -> 8 tiles per wave exactly

// per-wave LDS layout
#define A_ROW_BYTES 144                          // 64 halves + 8 pad halves (bank-friendly, 16B aligned)
#define A_MAT_BYTES (ROWS_PER_TILE * A_ROW_BYTES)  // 4608
#define Z_BYTES 256
#define WAVE_LDS (Z_BYTES + 3 * A_MAT_BYTES)       // 14080

#define LOG2E 1.44269504088896340736f
#define LN2   0.69314718055994530942f

// softplus + sigmoid from one raw v_exp_f32 / v_log_f32 / v_rcp_f32.
// Argument of exp2 is <= 0 (no overflow); argument of log2 is in [1,2]
// (never denormal), so the libm range-fixup code is unnecessary.
__device__ __forceinline__ void sp_sig(float p, float& sp, float& sg) {
    float en = __builtin_amdgcn_exp2f(-fabsf(p) * LOG2E);   // exp(-|p|)
    float r  = __builtin_amdgcn_rcpf(1.0f + en);
    sp = fmaxf(p, 0.0f) + LN2 * __builtin_amdgcn_logf(1.0f + en);
    sg = (p >= 0.0f) ? r : en * r;
}

__device__ __forceinline__ v8f wmma32(v16h a, v16h b, v8f c) {
    // D = A(16x32 f16) * B(32x16 f16) + C(16x16 f32)
    return __builtin_amdgcn_wmma_f32_16x16x32_f16(false, a, false, b, (short)0, c, false, false);
}

__global__ __launch_bounds__(BLOCK) void cnf_odefunc_kernel(
    const float* __restrict__ T,  const float* __restrict__ Z,
    const float* __restrict__ W0, const float* __restrict__ b0,
    const float* __restrict__ W1, const float* __restrict__ b1,
    const float* __restrict__ W2, const float* __restrict__ b2,
    float* __restrict__ out)
{
    __shared__ __align__(16) unsigned char smem[WAVES_PER_BLOCK * WAVE_LDS];

    const int tid  = threadIdx.x;
    const int wave = tid >> 5;
    const int lane = tid & 31;
    const int hw   = lane >> 4;     // half-wave id (0/1)
    const int ln   = lane & 15;

    unsigned char* wbase = smem + wave * WAVE_LDS;
    float2* zb = (float2*)wbase;
    unsigned char* ab = wbase + Z_BYTES;

    const float t = T[0];

    // ---- resident B fragments for W1' (64x64, K-major halves per column lane) ----
    v16h bfrag[4][2];
    #pragma unroll
    for (int nt = 0; nt < 4; ++nt) {
      #pragma unroll
      for (int kc = 0; kc < 2; ++kc) {
        const int N  = nt * 16 + ln;
        const int Kb = kc * 32 + hw * 16;
        v8u u;
        #pragma unroll
        for (int j = 0; j < 8; ++j) {
          float f0 = W1[(1 + Kb + 2*j    ) * HID + N];
          float f1 = W1[(1 + Kb + 2*j + 1) * HID + N];
          v2h ph = __builtin_amdgcn_cvt_pkrtz(f0, f1);
          u[j] = __builtin_bit_cast(unsigned int, ph);
        }
        bfrag[nt][kc] = __builtin_bit_cast(v16h, u);
      }
    }

    // ---- per-lane layer-0 constants: lane owns hidden units c0=2*lane, c1=2*lane+1 ----
    const int c0 = 2 * lane, c1 = 2 * lane + 1;
    const float w0a0 = W0[64 + c0], w0b0 = W0[128 + c0];
    const float w0a1 = W0[64 + c1], w0b1 = W0[128 + c1];
    const float b0p0 = b0[c0] + t * W0[c0];       // fold t row into bias
    const float b0p1 = b0[c1] + t * W0[c1];

    // ---- per-lane epilogue constants (WMMA C column = nt*16 + ln) ----
    float b1p[4], w20[4], w21[4];
    #pragma unroll
    for (int nt = 0; nt < 4; ++nt) {
      const int col = nt * 16 + ln;
      b1p[nt] = b1[col] + t * W1[col];            // W1 row 0 is the t row
      w20[nt] = W2[(1 + col) * 2 + 0];
      w21[nt] = W2[(1 + col) * 2 + 1];
    }
    const float b2p0 = b2[0] + t * W2[0];
    const float b2p1 = b2[1] + t * W2[1];

    const int waveGlobal = blockIdx.x * WAVES_PER_BLOCK + wave;
    float* outdiv = out + 2 * BATCH;

    for (int tile = waveGlobal; tile < NTILES; tile += TOTAL_WAVES) {
      const int base = tile * ROWS_PER_TILE;

      // stage z tile (coalesced b64), wave-private LDS => no barrier, only DS wait
      zb[lane] = ((const float2*)Z)[base + lane];
      asm volatile("s_wait_dscnt 0" ::: "memory");

      // ---- layer 0: build A matrices h0, dh0/dz0, dh0/dz1 (f16), c-sliced across lanes ----
      #pragma unroll 8
      for (int r = 0; r < ROWS_PER_TILE; ++r) {
        float2 zr = zb[r];                         // LDS broadcast read
        float p0 = b0p0 + zr.x * w0a0 + zr.y * w0b0;
        float p1 = b0p1 + zr.x * w0a1 + zr.y * w0b1;
        float sp0, sg0, sp1, sg1;
        sp_sig(p0, sp0, sg0);
        sp_sig(p1, sp1, sg1);
        unsigned char* rowp = ab + r * A_ROW_BYTES + lane * 4;
        *(v2h*)(rowp)                 = __builtin_amdgcn_cvt_pkrtz(sp0, sp1);
        *(v2h*)(rowp + A_MAT_BYTES)   = __builtin_amdgcn_cvt_pkrtz(sg0 * w0a0, sg1 * w0a1);
        *(v2h*)(rowp + 2*A_MAT_BYTES) = __builtin_amdgcn_cvt_pkrtz(sg0 * w0b0, sg1 * w0b1);
      }
      asm volatile("s_wait_dscnt 0" ::: "memory");

      #pragma unroll
      for (int mt = 0; mt < 2; ++mt) {
        // gather A fragments in contiguous K order (same as bfrag):
        // lane 0-15 row=ln holds K=Kb..Kb+15 ; lanes 16-31 Kb+=16
        v16h afrag[3][2];
        const int arow = mt * 16 + ln;
        #pragma unroll
        for (int m = 0; m < 3; ++m) {
          #pragma unroll
          for (int kc = 0; kc < 2; ++kc) {
            const int Kb = kc * 32 + hw * 16;
            const unsigned char* p = ab + m * A_MAT_BYTES + arow * A_ROW_BYTES + Kb * 2;
            v4u lo = *(const v4u*)p;           // halves 0..7
            v4u hi = *(const v4u*)(p + 16);    // halves 8..15 (K+8)
            v8u u = { lo.x, lo.y, lo.z, lo.w, hi.x, hi.y, hi.z, hi.w };
            afrag[m][kc] = __builtin_bit_cast(v16h, u);
          }
        }

        float zd0[8], zd1[8], dvv[8];
        #pragma unroll
        for (int v = 0; v < 8; ++v) { zd0[v] = 0.f; zd1[v] = 0.f; dvv[v] = 0.f; }

        #pragma unroll
        for (int nt = 0; nt < 4; ++nt) {
          v8f cz = {};
          v8f ch  = wmma32(afrag[0][0], bfrag[nt][0], cz);
          ch      = wmma32(afrag[0][1], bfrag[nt][1], ch);
          v8f cd0 = wmma32(afrag[1][0], bfrag[nt][0], cz);
          cd0     = wmma32(afrag[1][1], bfrag[nt][1], cd0);
          v8f cd1 = wmma32(afrag[2][0], bfrag[nt][0], cz);
          cd1     = wmma32(afrag[2][1], bfrag[nt][1], cd1);

          // epilogue directly in C layout: lane holds col=nt*16+ln, rows v+8*hw
          #pragma unroll
          for (int v = 0; v < 8; ++v) {
            float p = ch[v] + b1p[nt];
            float sp, sg;
            sp_sig(p, sp, sg);
            zd0[v] += sp * w20[nt];
            zd1[v] += sp * w21[nt];
            dvv[v] += sg * (cd0[v] * w20[nt] + cd1[v] * w21[nt]);
          }
        }

        // reduce over the 16 column-lanes of each half-wave
        #pragma unroll
        for (int v = 0; v < 8; ++v) {
          #pragma unroll
          for (int mask = 1; mask < 16; mask <<= 1) {
            zd0[v] += __shfl_xor(zd0[v], mask, 32);
            zd1[v] += __shfl_xor(zd1[v], mask, 32);
            dvv[v] += __shfl_xor(dvv[v], mask, 32);
          }
        }

        if (ln < 8) {
          const int v = lane & 7;
          float o0 = zd0[0], o1 = zd1[0], od = dvv[0];
          #pragma unroll
          for (int k = 1; k < 8; ++k)
            if (v == k) { o0 = zd0[k]; o1 = zd1[k]; od = dvv[k]; }
          const int row = base + mt * 16 + hw * 8 + v;
          ((float2*)out)[row] = make_float2(o0 + b2p0, o1 + b2p1);  // z_dot
          outdiv[row] = -od;                                        // -divergence
        }
      }
    }
}

extern "C" void kernel_launch(void* const* d_in, const int* in_sizes, int n_in,
                              void* d_out, int out_size, void* d_ws, size_t ws_size,
                              hipStream_t stream) {
    // setup_inputs order: t, z, delta_logpz, W0, b0, W1, b1, W2, b2
    const float* T  = (const float*)d_in[0];
    const float* Z  = (const float*)d_in[1];
    // d_in[2] = delta_logpz (unused by reference math)
    const float* W0 = (const float*)d_in[3];
    const float* b0 = (const float*)d_in[4];
    const float* W1 = (const float*)d_in[5];
    const float* b1 = (const float*)d_in[6];
    const float* W2 = (const float*)d_in[7];
    const float* b2 = (const float*)d_in[8];
    float* out = (float*)d_out;  // [z_dot (B,2) | -div (B,1)] flat

    cnf_odefunc_kernel<<<GRID, BLOCK, 0, stream>>>(T, Z, W0, b0, W1, b1, W2, b2, out);
}
